// GlobalInformationModule_4715874091244
// MI455X (gfx1250) — matrix operand order repaired and checked by reading, not compiled, not run on hardware
//
#include <hip/hip_runtime.h>
#include <hip/hip_bf16.h>
#include <math.h>

typedef __attribute__((ext_vector_type(16))) __bf16 v16bf;
typedef __attribute__((ext_vector_type(8)))  float  v8f;

#define Pn 224
#define Ln 350
#define Dn 768
#define MT 352          // L padded to 16
#define KSTEPS 24       // 768 / 32
#define NTILES 48       // 768 / 16

// workspace layout (bytes)
#define W1T_OFF 0
#define W1T_BYTES (768*768*2)          // 1,179,648
#define EMB_OFF  W1T_BYTES             // 448*768*4 = 1,376,256
#define SC_OFF   (EMB_OFF + 448*768*4) // 448*4

// ---------------------------------------------------------------------------
// Pack W1 (row-major K x N, f32) into bf16 WMMA-B tiles.
// B 32x16 bf16 order chosen here (contiguous k, same as A staging): lane t (h=t/16, i=t%16) holds column N=i,
// K = 16*h + 2*r + j for VGPR r, half j. Tile = 32 lanes * 16 elems,
// stored contiguously so each lane loads one aligned 32-byte v16bf.
// ---------------------------------------------------------------------------
__global__ void __launch_bounds__(256) k_w1_to_bf16_tiles(
    const float* __restrict__ W1, __bf16* __restrict__ w1t)
{
    int o    = blockIdx.x * 256 + threadIdx.x;   // 0 .. 589823
    int tile = o >> 9;          // 512 elems / tile
    int e    = o & 511;
    int t    = e >> 4;          // lane
    int q    = e & 15;          // element within lane
    int r    = q >> 1;
    int j    = q & 1;
    int nb   = tile / KSTEPS;
    int kb   = tile % KSTEPS;
    int h    = t >> 4;
    int i    = t & 15;
    int K    = kb * 32 + 16 * h + 2 * r + j;
    int N    = nb * 16 + i;
    w1t[o] = (__bf16)W1[(size_t)K * Dn + N];
}

// ---------------------------------------------------------------------------
// Fused per-pair kernel: scores = tanh(X[p] @ W1 + b1) @ v1  (WMMA bf16),
// then two masked softmax poolings over L re-using X[p] (L2-resident),
// plus the pair_embeddings copy (row 0).
// ---------------------------------------------------------------------------
__global__ void __launch_bounds__(256) k_scores_pool(
    const float* __restrict__ X, const int* __restrict__ fsep,
    const int* __restrict__ ssep, const __bf16* __restrict__ w1t,
    const float* __restrict__ b1, const float* __restrict__ v1,
    float* __restrict__ emb, float* __restrict__ out_pair)
{
    __shared__ alignas(32) __bf16 a_lds[16 * Dn];   // 24 KB: one 16-row A slab
    __shared__ float scores[MT];

    const int p    = blockIdx.x;
    const int tid  = threadIdx.x;
    const int lane = tid & 31;
    const int wv   = tid >> 5;
    const int h    = lane >> 4;     // lane half
    const int i    = lane & 15;     // lane within half

    for (int f = tid; f < MT; f += 256) scores[f] = 0.f;
    __syncthreads();

    for (int m0 = 0; m0 < MT; m0 += 16) {
        // stage 16 rows of X as bf16 in 16x32 A tiles, same contiguous k order as packed B:
        // lane t (h=t/16, i=t%16) row M=i, K = 16h + 2r + j
        for (int f = tid; f < 16 * Dn; f += 256) {
            int kb  = f >> 9;
            int rem = f & 511;
            int t   = rem >> 4;
            int q   = rem & 15;
            int r   = q >> 1;
            int j   = q & 1;
            int th  = t >> 4;
            int ti  = t & 15;
            int K   = kb * 32 + 16 * th + 2 * r + j;
            int row = m0 + ti;
            float v = (row < Ln) ? X[((size_t)p * Ln + row) * Dn + K] : 0.f;
            a_lds[f] = (__bf16)v;
        }
        __syncthreads();

        // per-wave partial score for rows m0 + r + 8h, accumulated in
        // registers across this wave's 6 N-tiles (no LDS atomics in the loop)
        float ssum[8];
#pragma unroll
        for (int r = 0; r < 8; ++r) ssum[r] = 0.f;

        for (int step = 0; step < 6; ++step) {
            int nb = wv + 8 * step;
            v8f acc = {};
            const __bf16* bbase = w1t + (size_t)nb * KSTEPS * 512;
            for (int kb = 0; kb < KSTEPS; ++kb) {
                v16bf a = *(const v16bf*)(a_lds + ((kb << 5) + lane) * 16);
                v16bf b = *(const v16bf*)(bbase + (size_t)((kb << 5) + lane) * 16);
                acc = __builtin_amdgcn_wmma_f32_16x16x32_bf16(
                        false, a, false, b, (short)0, acc, false, false);
            }
            // C/D 16x16 f32: VGPR r, lane (h,i) -> element (M = r + 8h, N = i)
            int   N   = nb * 16 + i;
            float b1n = b1[N];
            float v1n = v1[N];
#pragma unroll
            for (int r = 0; r < 8; ++r)
                ssum[r] += tanhf(acc[r] + b1n) * v1n;
        }

        // butterfly-reduce over the 16 lanes of each half (masks stay inside
        // the half), then one ds_add_f32 per row per wave
#pragma unroll
        for (int r = 0; r < 8; ++r) {
            float v = ssum[r];
            v += __shfl_xor(v, 1);
            v += __shfl_xor(v, 2);
            v += __shfl_xor(v, 4);
            v += __shfl_xor(v, 8);
            int M = m0 + r + 8 * h;
            if (i == 0 && M < Ln)
                atomicAdd(&scores[M], v);
        }
        __syncthreads();
    }

    // -------- masked softmax pooling (vb1 cancels in softmax) --------
    int fs = fsep[p], ss = ssep[p];
    float mx1 = -3.0e38f, mx2 = -3.0e38f;
    for (int l = 1; l < fs; ++l)       mx1 = fmaxf(mx1, scores[l]);
    for (int l = fs + 1; l < ss; ++l)  mx2 = fmaxf(mx2, scores[l]);
    float dn1 = 0.f, dn2 = 0.f;
    for (int l = 1; l < fs; ++l)       dn1 += expf(scores[l] - mx1);
    for (int l = fs + 1; l < ss; ++l)  dn2 += expf(scores[l] - mx2);

    float* wpool = (float*)a_lds;   // reuse LDS: 2*MT floats
    for (int l = tid; l < MT; l += 256) {
        float s = scores[l];
        wpool[l]      = (l >= 1 && l < fs)     ? expf(s - mx1) / dn1 : 0.f;
        wpool[MT + l] = (l > fs && l < ss)     ? expf(s - mx2) / dn2 : 0.f;
    }
    __syncthreads();

    for (int c = 0; c < 3; ++c) {
        int d = tid + (c << 8);
        float a1 = 0.f, a2 = 0.f;
        for (int l = 1; l < ss; ++l) {
            float xv = X[((size_t)p * Ln + l) * Dn + d];   // L2 hit
            a1 += wpool[l] * xv;
            a2 += wpool[MT + l] * xv;
        }
        emb[((size_t)(2 * p)) * Dn + d]     = a1;
        emb[((size_t)(2 * p + 1)) * Dn + d] = a2;
        out_pair[(size_t)p * Dn + d]        = X[((size_t)p * Ln) * Dn + d];
    }
}

// ---------------------------------------------------------------------------
// Second head: sc = tanh(emb @ W2 + b2) @ v2   (tiny: 0.5 GFLOP)
// ---------------------------------------------------------------------------
__global__ void __launch_bounds__(256) k_head2(
    const float* __restrict__ emb, const float* __restrict__ W2,
    const float* __restrict__ b2, const float* __restrict__ v2,
    float* __restrict__ sc)
{
    __shared__ float row[Dn];
    __shared__ float red[256];
    int rix = blockIdx.x;
    int tid = threadIdx.x;
    for (int k = tid; k < Dn; k += 256) row[k] = emb[(size_t)rix * Dn + k];
    __syncthreads();
    float part = 0.f;
    for (int c = 0; c < 3; ++c) {
        int n = tid + (c << 8);
        float dot = 0.f;
        for (int k = 0; k < Dn; ++k) dot += row[k] * W2[(size_t)k * Dn + n];
        part += tanhf(dot + b2[n]) * v2[n];
    }
    red[tid] = part;
    __syncthreads();
    for (int st = 128; st > 0; st >>= 1) {
        if (tid < st) red[tid] += red[tid + st];
        __syncthreads();
    }
    if (tid == 0) sc[rix] = red[0];
}

// ---------------------------------------------------------------------------
// Segment softmax pool: one block per segment (z,j); 14 members derived
// analytically from the pair enumeration (z-major, j, k skipping j).
// vb2 cancels in the segment softmax.
// ---------------------------------------------------------------------------
__global__ void __launch_bounds__(256) k_segment(
    const float* __restrict__ emb, const float* __restrict__ sc,
    float* __restrict__ out2)
{
    __shared__ int   mem[14];
    __shared__ float wgt[14];
    int s   = blockIdx.x;
    int z   = s >> 3;
    int j   = s & 7;
    int tid = threadIdx.x;
    if (tid < 14) {
        int c = (tid < 7) ? tid : tid - 7;
        int k = c + (c >= j ? 1 : 0);
        int m;
        if (tid < 7) {
            m = 2 * (z * 56 + j * 7 + c);              // e1 of pair (z,j,k)
        } else {
            int jj = j - (j > k ? 1 : 0);
            m = 2 * (z * 56 + k * 7 + jj) + 1;          // e2 of pair (z,k,j)
        }
        mem[tid] = m;
    }
    __syncthreads();
    if (tid == 0) {
        float mx = -3.0e38f;
        for (int m = 0; m < 14; ++m) mx = fmaxf(mx, sc[mem[m]]);
        float dn = 0.f;
        for (int m = 0; m < 14; ++m) {
            float e = expf(sc[mem[m]] - mx);
            wgt[m] = e; dn += e;
        }
        float inv = 1.f / dn;
        for (int m = 0; m < 14; ++m) wgt[m] *= inv;
    }
    __syncthreads();
    for (int c = 0; c < 3; ++c) {
        int d = tid + (c << 8);
        float a = 0.f;
        for (int m = 0; m < 14; ++m)
            a += wgt[m] * emb[(size_t)mem[m] * Dn + d];
        out2[(size_t)s * Dn + d] = a;
    }
}

// ---------------------------------------------------------------------------
extern "C" void kernel_launch(void* const* d_in, const int* in_sizes, int n_in,
                              void* d_out, int out_size, void* d_ws, size_t ws_size,
                              hipStream_t stream)
{
    const float* X  = (const float*)d_in[0];
    const int*   fs = (const int*)d_in[1];
    const int*   ss = (const int*)d_in[2];
    const float* W1 = (const float*)d_in[3];
    const float* b1 = (const float*)d_in[4];
    const float* v1 = (const float*)d_in[5];
    // d_in[6] = vb1 : additive constant inside softmax -> cancels, unused
    const float* W2 = (const float*)d_in[7];
    const float* b2 = (const float*)d_in[8];
    const float* v2 = (const float*)d_in[9];
    // d_in[10] = vb2 : cancels in segment softmax, unused
    (void)in_sizes; (void)n_in; (void)out_size; (void)ws_size;

    char*   ws  = (char*)d_ws;
    __bf16* w1t = (__bf16*)(ws + W1T_OFF);
    float*  emb = (float*)(ws + EMB_OFF);
    float*  scv = (float*)(ws + SC_OFF);

    float* out_pair = (float*)d_out;                       // (P, D)
    float* out2     = (float*)d_out + (size_t)Pn * Dn;     // (B, S, D)

    k_w1_to_bf16_tiles<<<(768 * 768) / 256, 256, 0, stream>>>(W1, w1t);
    k_scores_pool<<<Pn, 256, 0, stream>>>(X, fs, ss, w1t, b1, v1, emb, out_pair);
    k_head2<<<448, 256, 0, stream>>>(emb, W2, b2, v2, scv);
    k_segment<<<32, 256, 0, stream>>>(emb, scv, out2);
}
